// MeshDeform_65000035058095
// MI455X (gfx1250) — matrix operand order repaired and checked by reading, not compiled, not run on hardware
//
#include <hip/hip_runtime.h>
#include <hip/hip_bf16.h>
#include <stdint.h>

// ---------------------------------------------------------------------------
// CDNA5 (gfx1250, wave32) implementation.
// Conv layers: implicit GEMM on V_WMMA_F32_16X16X32_BF16 (bf16 in, fp32 acc),
//   64 voxels x 32 channels per wave, weights staged per td-plane in LDS,
//   unsigned 32-bit offsets (saddr-form loads), halo via zeroed buffer tails.
// Integration / warp / gaussian / vertex-interp: fp32 VALU (bandwidth bound).
// ---------------------------------------------------------------------------

typedef __attribute__((ext_vector_type(16))) __bf16 v16bf;
typedef __attribute__((ext_vector_type(8)))  __bf16 v8bf;
typedef __attribute__((ext_vector_type(8)))  float  v8f;

#define LRELU_SLOPE 0.2f

// ============================ weight / bias prep ===========================
// Re-swizzle conv weights (Cout, Cin, 3,3,3) fp32 -> bf16 fragments laid out
// exactly as each lane of a wave32 WMMA B-operand consumes them:
//   frag f = ((t*kcnt + kc)*2 + nb), element addr = (f*32 + lane)*16 + e
//   lane l supplies N = (l&15)+nb*16,  K = (e&7) + (e>>3)*16 + (l<16 ? 0 : 8) within kc*32.
__global__ void prep_weights_kernel(const float* __restrict__ w,
                                    __bf16* __restrict__ out,
                                    int CinA, int CoutA, int kcnt)
{
    int idx = blockIdx.x * blockDim.x + threadIdx.x;
    int total = 27 * kcnt * 2 * 512;
    if (idx >= total) return;
    int e   = idx & 15;
    int l   = (idx >> 4) & 31;
    int nb  = (idx >> 9) & 1;
    int rest = idx >> 10;
    int kc  = rest % kcnt;
    int t   = rest / kcnt;
    int n   = (l & 15) + nb * 16;
    int K   = (e & 7) + ((e >> 3) << 4) + ((l < 16) ? 0 : 8);
    int ci  = kc * 32 + K;
    float v = 0.0f;
    if (ci < CinA && n < CoutA)
        v = w[((size_t)n * CinA + ci) * 27 + t];
    out[idx] = (__bf16)v;
}

__global__ void prep_bias_kernel(const float* __restrict__ b,
                                 float* __restrict__ out, int CoutA)
{
    int c = threadIdx.x;
    if (c < 32) out[c] = (c < CoutA) ? b[c] : 0.0f;
}

__global__ void zero_bf16_kernel(__bf16* __restrict__ p, int n)
{
    int i = blockIdx.x * blockDim.x + threadIdx.x;
    if (i < n) p[i] = (__bf16)0.0f;
}

// Pack fp32 single-channel volume into channels-last bf16, channels padded to 32.
__global__ void pack_input_kernel(const float* __restrict__ vol,
                                  __bf16* __restrict__ dst, long nv)
{
    long idx = (long)blockIdx.x * blockDim.x + threadIdx.x;
    if (idx >= nv * 32) return;
    int  c = (int)(idx & 31);
    long v = idx >> 5;
    dst[idx] = (c == 0) ? (__bf16)vol[v] : (__bf16)0.0f;
}

// ====================== WMMA implicit-GEMM 3x3x3 conv ======================
// One wave computes 64 output voxels x 32 output channels (4 M-subtiles, two
// N-halves). One td-plane of weight fragments (9 taps) is staged global->LDS
// per block (ds_load_b128 readback); each B pair feeds 8 WMMAs. All 36
// predicated A offsets for the plane are precomputed as unsigned 32-bit
// values (khalf folded in) so loads select the saddr (SGPR base + zext VGPR
// offset) form with pure immediate sub-offsets. Out-of-halo lanes read the
// buffer's zeroed 256B tail (offset zoff) via one 32-bit cndmask.
template<int KCNT, int SC>
__global__ __launch_bounds__(128)
void conv3_wmma_kernel(const __bf16* __restrict__ src, __bf16* __restrict__ dst,
                       const __bf16* __restrict__ wgt, const float* __restrict__ bias,
                       int D, int H, int W, int Do, int Ho, int Wo,
                       int sO, int dC, int dO, int stride, unsigned zoff)
{
    __shared__ __align__(16) __bf16 smem[9 * KCNT * 2 * 512];   // one td-plane

    const int lane  = threadIdx.x & 31;
    const int wave  = threadIdx.x >> 5;
    const int nvox  = Do * Ho * Wo;
    const int m0    = (blockIdx.x * 4 + wave) * 64;
    const int row   = lane & 15;
    const int khalf = (lane < 16) ? 0 : 8;  // A-layout channel sub-offset per lane half
    const unsigned zoffk = zoff + (unsigned)khalf;

    int  id0[4], ih0[4], iw0[4];
    bool mval[4];
    bool wok[4][3];
    #pragma unroll
    for (int j = 0; j < 4; ++j) {
        int mr  = m0 + j * 16 + row;
        mval[j] = mr < nvox;
        int mc  = mval[j] ? mr : 0;
        int od  = mc / (Ho * Wo);
        int r   = mc % (Ho * Wo);
        int oh  = r / Wo, ow = r % Wo;
        id0[j] = od * stride - 1;
        ih0[j] = oh * stride - 1;
        iw0[j] = ow * stride - 1;
        #pragma unroll
        for (int tw = 0; tw < 3; ++tw)
            wok[j][tw] = (unsigned)(iw0[j] + tw) < (unsigned)W;
    }

    v8f acc[4][2];
    #pragma unroll
    for (int j = 0; j < 4; ++j) { acc[j][0] = {}; acc[j][1] = {}; }

    for (int td = 0; td < 3; ++td) {
        __syncthreads();
        {   // cooperative stage of this td-plane's B fragments into LDS
            const uint4* gsrc = (const uint4*)(wgt + (size_t)td * 9 * KCNT * 1024);
            uint4* sdst = (uint4*)smem;
            #pragma unroll 1
            for (int i = threadIdx.x; i < 9 * KCNT * 128; i += 128) sdst[i] = gsrc[i];
        }
        __syncthreads();
        if (td < 2)                                      // warm next plane
            __builtin_prefetch(wgt + (size_t)(td + 1) * 9 * KCNT * 1024, 0, 1);

        // ---- precompute all 36 predicated u32 A offsets for this plane ----
        unsigned eoff[4][9];
        #pragma unroll
        for (int j = 0; j < 4; ++j) {
            const int  idv  = id0[j] + td;
            const bool dok  = mval[j] && ((unsigned)idv < (unsigned)D);
            const int  drow = (idv < 0 ? 0 : idv) * H;
            #pragma unroll
            for (int th = 0; th < 3; ++th) {
                const int  ihv = ih0[j] + th;
                const bool rok = dok && ((unsigned)ihv < (unsigned)H);
                const int  ihc = (ihv < 0) ? 0 : ihv;
                const int  iwc = (iw0[j] < 0) ? 0 : iw0[j];
                const unsigned rowoff =
                    (unsigned)(((drow + ihc) * W + iwc) * SC + sO + khalf
                               - ((iw0[j] < 0) ? SC : 0));   // exact iw0 offset
                #pragma unroll
                for (int tw = 0; tw < 3; ++tw)
                    eoff[j][th * 3 + tw] =
                        (rok && wok[j][tw]) ? (rowoff + (unsigned)(tw * SC)) : zoffk;
            }
        }

        // ---- pure load->wmma sweep over the 9 taps of this plane ----
        #pragma unroll
        for (int tt = 0; tt < 9; ++tt) {
            #pragma unroll
            for (int kc = 0; kc < KCNT; ++kc) {
                const __bf16* bf = smem + ((tt * KCNT + kc) * 2) * 512 + lane * 16;
                const v16bf b0 = *(const v16bf*)(bf);
                const v16bf b1 = *(const v16bf*)(bf + 512);
                #pragma unroll
                for (int j = 0; j < 4; ++j) {
                    const __bf16* sp = src + (size_t)eoff[j][tt];  // zext -> saddr form
                    const v8bf lo = *(const v8bf*)(sp + kc * 32);
                    const v8bf hi = *(const v8bf*)(sp + kc * 32 + 16);
                    v16bf a;
                    #pragma unroll
                    for (int e = 0; e < 8; ++e) { a[e] = lo[e]; a[8 + e] = hi[e]; }
                    acc[j][0] = __builtin_amdgcn_wmma_f32_16x16x32_bf16(
                        false, a, false, b0, (short)0, acc[j][0], false, false);
                    acc[j][1] = __builtin_amdgcn_wmma_f32_16x16x32_bf16(
                        false, a, false, b1, (short)0, acc[j][1], false, false);
                }
            }
        }
    }

    // Epilogue: C/D layout -> lane l holds N=(l&15), rows g (+8 for upper half).
    // Lanes 0..15 of each store cover 16 consecutive channels => coalesced.
    const int   n    = lane & 15;
    const float b0v  = bias[n];
    const float b1v  = bias[n + 16];
    const int   moff = (lane < 16) ? 0 : 8;
    #pragma unroll
    for (int j = 0; j < 4; ++j) {
        #pragma unroll
        for (int gg = 0; gg < 8; ++gg) {
            const int mm = m0 + j * 16 + gg + moff;
            if (mm < nvox) {
                float v0 = acc[j][0][gg] + b0v; v0 = (v0 > 0.f) ? v0 : v0 * LRELU_SLOPE;
                float v1 = acc[j][1][gg] + b1v; v1 = (v1 > 0.f) ? v1 : v1 * LRELU_SLOPE;
                const unsigned doff = (unsigned)(mm * dC + dO);  // u32 store offset
                dst[(size_t)doff + n]      = (__bf16)v0;
                dst[(size_t)doff + n + 16] = (__bf16)v1;
            }
        }
    }
}

// ======================= flow convs (Cout=3, fp32 out) =====================
__global__ __launch_bounds__(256)
void flow_conv_kernel(const __bf16* __restrict__ src, float* __restrict__ dst,
                      const float* __restrict__ w, const float* __restrict__ b,
                      int D, int H, int W, int Ctot, int Coff, int CinA)
{
    __shared__ float sw[3 * 32 * 27];
    const int nw = 3 * CinA * 27;
    for (int i = threadIdx.x; i < nw; i += blockDim.x) sw[i] = w[i];
    __syncthreads();

    long v = (long)blockIdx.x * blockDim.x + threadIdx.x;
    long nv = (long)D * H * W;
    if (v >= nv) return;
    int d = (int)(v / ((long)H * W));
    int r = (int)(v % ((long)H * W));
    int h = r / W, ww = r % W;
    float a0 = b[0], a1 = b[1], a2 = b[2];
    for (int t = 0; t < 27; ++t) {
        int td = t / 9, th = (t / 3) % 3, tw = t % 3;
        int id = d - 1 + td, ih = h - 1 + th, iw = ww - 1 + tw;
        if ((unsigned)id < (unsigned)D && (unsigned)ih < (unsigned)H &&
            (unsigned)iw < (unsigned)W) {
            const __bf16* sp = src + ((size_t)((id * H + ih) * W + iw) * Ctot + Coff);
            for (int ci = 0; ci < CinA; ++ci) {
                float x = (float)sp[ci];
                a0 += x * sw[(0 * CinA + ci) * 27 + t];
                a1 += x * sw[(1 * CinA + ci) * 27 + t];
                a2 += x * sw[(2 * CinA + ci) * 27 + t];
            }
        }
    }
    dst[v] = a0; dst[nv + v] = a1; dst[2 * nv + v] = a2;
}

// ============================ trilinear upsample ===========================
__device__ __forceinline__ void up_coord(int o, int n, int& i0, int& i1, float& w1)
{
    float s = (o + 0.5f) * 0.5f - 0.5f;
    s = fmaxf(s, 0.0f);
    i0 = (int)floorf(s);
    if (i0 > n - 1) i0 = n - 1;
    i1 = min(i0 + 1, n - 1);
    w1 = s - (float)i0;
}

__global__ void upsample_bf16_kernel(const __bf16* __restrict__ src, __bf16* __restrict__ dst,
                                     int Ds, int Hs, int Ws,
                                     int sC, int sO, int dC, int dO, int C)
{
    int Dd = 2 * Ds, Hd = 2 * Hs, Wd = 2 * Ws;
    long idx = (long)blockIdx.x * blockDim.x + threadIdx.x;
    long total = (long)Dd * Hd * Wd * C;
    if (idx >= total) return;
    int  c = (int)(idx % C);
    long v = idx / C;
    int od = (int)(v / ((long)Hd * Wd));
    int r  = (int)(v % ((long)Hd * Wd));
    int oh = r / Wd, ow = r % Wd;
    int d0, d1, h0, h1, w0, w1i; float fd, fh, fw;
    up_coord(od, Ds, d0, d1, fd);
    up_coord(oh, Hs, h0, h1, fh);
    up_coord(ow, Ws, w0, w1i, fw);
    auto S = [&](int dd, int hh, int wv) -> float {
        return (float)src[(size_t)((dd * Hs + hh) * Ws + wv) * sC + sO + c];
    };
    float r00 = S(d0, h0, w0) * (1.f - fw) + S(d0, h0, w1i) * fw;
    float r01 = S(d0, h1, w0) * (1.f - fw) + S(d0, h1, w1i) * fw;
    float r10 = S(d1, h0, w0) * (1.f - fw) + S(d1, h0, w1i) * fw;
    float r11 = S(d1, h1, w0) * (1.f - fw) + S(d1, h1, w1i) * fw;
    float q0  = r00 * (1.f - fh) + r01 * fh;
    float q1  = r10 * (1.f - fh) + r11 * fh;
    dst[(size_t)((od * Hd + oh) * Wd + ow) * dC + dO + c] = (__bf16)(q0 * (1.f - fd) + q1 * fd);
}

__global__ void upsample3_f32_kernel(const float* __restrict__ src, float* __restrict__ dst,
                                     int Ds, int Hs, int Ws)
{
    int Dd = 2 * Ds, Hd = 2 * Hs, Wd = 2 * Ws;
    long nvd = (long)Dd * Hd * Wd, nvs = (long)Ds * Hs * Ws;
    long idx = (long)blockIdx.x * blockDim.x + threadIdx.x;
    if (idx >= 3 * nvd) return;
    int  c = (int)(idx / nvd);
    long v = idx % nvd;
    int od = (int)(v / ((long)Hd * Wd));
    int r  = (int)(v % ((long)Hd * Wd));
    int oh = r / Wd, ow = r % Wd;
    int d0, d1, h0, h1, w0, w1i; float fd, fh, fw;
    up_coord(od, Ds, d0, d1, fd);
    up_coord(oh, Hs, h0, h1, fh);
    up_coord(ow, Ws, w0, w1i, fw);
    const float* sp = src + (size_t)c * nvs;
    auto S = [&](int dd, int hh, int wv) -> float {
        return sp[((long)dd * Hs + hh) * Ws + wv];
    };
    float r00 = S(d0, h0, w0) * (1.f - fw) + S(d0, h0, w1i) * fw;
    float r01 = S(d0, h1, w0) * (1.f - fw) + S(d0, h1, w1i) * fw;
    float r10 = S(d1, h0, w0) * (1.f - fw) + S(d1, h0, w1i) * fw;
    float r11 = S(d1, h1, w0) * (1.f - fw) + S(d1, h1, w1i) * fw;
    float q0  = r00 * (1.f - fh) + r01 * fh;
    float q1  = r10 * (1.f - fh) + r11 * fh;
    dst[idx] = q0 * (1.f - fd) + q1 * fd;
}

// ===================== scaling-and-squaring integration ====================
__global__ void scale_kernel(const float* __restrict__ s, float* __restrict__ d,
                             long n, float f)
{
    long i = (long)blockIdx.x * blockDim.x + threadIdx.x;
    if (i < n) d[i] = s[i] * f;
}

// out = flow + warp(flow, flow, grid) : fused composition step
__global__ void warp_step_kernel(const float* __restrict__ flow, float* __restrict__ out,
                                 int D, int H, int W)
{
    long v = (long)blockIdx.x * blockDim.x + threadIdx.x;
    long nv = (long)D * H * W;
    if (v >= nv) return;
    int d = (int)(v / ((long)H * W));
    int r = (int)(v % ((long)H * W));
    int h = r / W, w = r % W;
    float f0 = flow[v], f1 = flow[nv + v], f2 = flow[2 * nv + v];
    float px = d + f0, py = h + f1, pz = w + f2;
    float fx = floorf(px), fy = floorf(py), fz = floorf(pz);
    int ix = (int)fx, iy = (int)fy, iz = (int)fz;
    float ux = px - fx, uy = py - fy, uz = pz - fz;
    float a0 = f0, a1 = f1, a2 = f2;
    #pragma unroll
    for (int dx = 0; dx < 2; ++dx)
    #pragma unroll
    for (int dy = 0; dy < 2; ++dy)
    #pragma unroll
    for (int dz = 0; dz < 2; ++dz) {
        int jx = ix + dx, jy = iy + dy, jz = iz + dz;
        bool inb = (jx >= 0) && (jx < D) && (jy >= 0) && (jy < H) && (jz >= 0) && (jz < W);
        float wt = (dx ? ux : 1.f - ux) * (dy ? uy : 1.f - uy) * (dz ? uz : 1.f - uz);
        if (inb) {
            long idx = ((long)jx * H + jy) * W + jz;
            a0 += flow[idx] * wt;
            a1 += flow[nv + idx] * wt;
            a2 += flow[2 * nv + idx] * wt;
        }
    }
    out[v] = a0; out[nv + v] = a1; out[2 * nv + v] = a2;
}

// depthwise 3x3x3 gaussian, sigma=0.5, zero-padded
__global__ void gauss3_kernel(const float* __restrict__ src, float* __restrict__ dst,
                              int D, int H, int W)
{
    long nv = (long)D * H * W;
    long idx = (long)blockIdx.x * blockDim.x + threadIdx.x;
    if (idx >= 3 * nv) return;
    int  c = (int)(idx / nv);
    long v = idx % nv;
    int d = (int)(v / ((long)H * W));
    int r = (int)(v % ((long)H * W));
    int h = r / W, w = r % W;
    const float e2 = 0.13533528323661270f;               // exp(-2)
    const float s1 = 1.0f + 2.0f * e2;
    const float w1d[3] = { e2 / s1, 1.0f / s1, e2 / s1 };
    const float* sp = src + (size_t)c * nv;
    float acc = 0.0f;
    #pragma unroll
    for (int td = 0; td < 3; ++td) {
        int id = d - 1 + td; if ((unsigned)id >= (unsigned)D) continue;
        #pragma unroll
        for (int th = 0; th < 3; ++th) {
            int ih = h - 1 + th; if ((unsigned)ih >= (unsigned)H) continue;
            #pragma unroll
            for (int tw = 0; tw < 3; ++tw) {
                int iw = w - 1 + tw; if ((unsigned)iw >= (unsigned)W) continue;
                acc += sp[((long)id * H + ih) * W + iw] * (w1d[td] * w1d[th] * w1d[tw]);
            }
        }
    }
    dst[idx] = acc;
}

// vert_out = vert_in + trilinear(phi at vert_in)
__global__ void interp_update_kernel(const float* __restrict__ vin,
                                     const float* __restrict__ phi,
                                     float* __restrict__ vout,
                                     int V, int D, int H, int W)
{
    int i = blockIdx.x * blockDim.x + threadIdx.x;
    if (i >= V) return;
    long nv = (long)D * H * W;
    float vx = vin[i * 3 + 0], vy = vin[i * 3 + 1], vz = vin[i * 3 + 2];
    float cx = fminf(fmaxf(vx, 1e-5f), (float)(D - 1) - 1e-5f);
    float cy = fminf(fmaxf(vy, 1e-5f), (float)(H - 1) - 1e-5f);
    float cz = fminf(fmaxf(vz, 1e-5f), (float)(W - 1) - 1e-5f);
    int ix = (int)floorf(cx), iy = (int)floorf(cy), iz = (int)floorf(cz);
    float ux = cx - ix, uy = cy - iy, uz = cz - iz;
    float a0 = 0.f, a1 = 0.f, a2 = 0.f;
    #pragma unroll
    for (int dx = 0; dx < 2; ++dx)
    #pragma unroll
    for (int dy = 0; dy < 2; ++dy)
    #pragma unroll
    for (int dz = 0; dz < 2; ++dz) {
        long idx = ((long)(ix + dx) * H + (iy + dy)) * W + (iz + dz);
        float wt = (dx ? ux : 1.f - ux) * (dy ? uy : 1.f - uy) * (dz ? uz : 1.f - uz);
        a0 += phi[idx] * wt;
        a1 += phi[nv + idx] * wt;
        a2 += phi[2 * nv + idx] * wt;
    }
    vout[i * 3 + 0] = vx + a0;
    vout[i * 3 + 1] = vy + a1;
    vout[i * 3 + 2] = vz + a2;
}

// ================================= host ====================================
extern "C" void kernel_launch(void* const* d_in, const int* in_sizes, int n_in,
                              void* d_out, int out_size, void* d_ws, size_t ws_size,
                              hipStream_t stream)
{
    const float* vert = (const float*)d_in[0];
    const float* vol  = (const float*)d_in[1];
    const float* cw[11]; const float* cb[11];
    for (int i = 0; i < 11; ++i) { cw[i] = (const float*)d_in[2 + 2 * i];
                                   cb[i] = (const float*)d_in[3 + 2 * i]; }
    const float* fw[4]; const float* fb[4];
    for (int i = 0; i < 4; ++i)  { fw[i] = (const float*)d_in[24 + 2 * i];
                                   fb[i] = (const float*)d_in[25 + 2 * i]; }
    const int NSTEPS = 7; // n_steps per setup_inputs (device scalar, fixed)

    const int D0 = 48, H0 = 96, W0 = 80; const long R0 = (long)D0 * H0 * W0;
    const int D1 = 24, H1 = 48, W1 = 40; const long R1 = (long)D1 * H1 * W1;
    const int D2 = 12, H2 = 24, W2 = 20; const long R2 = (long)D2 * H2 * W2;
    const int D3 =  6, H3 = 12, W3 = 10; const long R3 = (long)D3 * H3 * W3;
    const int NV = 150000;

    // conv metadata: conv1..conv6, deconv5..deconv1
    static const int CoutA[11] = {16,32,32,32,32,32, 32,32,32,32,16};
    static const int CinA [11] = { 1,16,32,32,32,32, 64,64,64,64,48};
    static const int CinP [11] = {32,32,32,32,32,32, 64,64,64,64,64};

    uint8_t* ws = (uint8_t*)d_ws;
    size_t off = 0;
    auto alloc = [&](size_t bytes) -> void* {
        void* p = ws + off;
        off += (bytes + 255) & ~(size_t)255;
        return p;
    };

    // bf16 activation buffers, each with a 256B zeroed tail (halo reads)
    auto allocA = [&](long elems) -> __bf16* {
        return (__bf16*)alloc((size_t)elems * 2 + 256);
    };
    const unsigned zIN  = (unsigned)(R0 * 32);
    const unsigned z64  = (unsigned)(R0 * 64);
    const unsigned zR1A = (unsigned)(R1 * 64), zR1B = (unsigned)(R1 * 32);
    const unsigned zR2A = (unsigned)(R2 * 64), zR2B = (unsigned)(R2 * 32);
    const unsigned zR3A = (unsigned)(R3 * 64), zR3B = (unsigned)(R3 * 32);
    __bf16* BUF_IN  = allocA(R0 * 32);   // packed input; later d1
    __bf16* BUF_T0A = allocA(R0 * 64);   // [up(d3) | x2]
    __bf16* BUF_T0B = allocA(R0 * 64);   // [d2     | x1(pad)]
    __bf16* BUF_R1A = allocA(R1 * 64);   // [up(d4) | x3]
    __bf16* BUF_R1B = allocA(R1 * 32);   // d3
    __bf16* BUF_R2A = allocA(R2 * 64);   // [up(d5) | x4]
    __bf16* BUF_R2B = allocA(R2 * 32);   // d4
    __bf16* BUF_R3A = allocA(R3 * 64);   // [x6     | x5]
    __bf16* BUF_R3B = allocA(R3 * 32);   // d5

    __bf16* WGT[11]; float* BIA[11];
    for (int i = 0; i < 11; ++i) {
        int kcnt = CinP[i] / 32;
        WGT[i] = (__bf16*)alloc((size_t)27 * kcnt * 1024 * 2);
        BIA[i] = (float*)alloc(32 * 4);
    }

    float* F_R2  = (float*)alloc(3 * R2 * 4);
    float* F_T1  = (float*)alloc(3 * R1 * 4);
    float* F_R1  = (float*)alloc(3 * R1 * 4);
    float* SVF[4];
    for (int i = 0; i < 4; ++i) SVF[i] = (float*)alloc(3 * R0 * 4);
    float* FLOWA = (float*)alloc(3 * R0 * 4);
    float* FLOWB = (float*)alloc(3 * R0 * 4);
    float* PHI   = (float*)alloc(3 * R0 * 4);
    float* VB0   = (float*)alloc((size_t)NV * 3 * 4);
    float* VB1   = (float*)alloc((size_t)NV * 3 * 4);
    (void)ws_size; (void)in_sizes; (void)n_in; (void)out_size;

    // ---- weight/bias prep, zero tails ----
    for (int i = 0; i < 11; ++i) {
        int kcnt = CinP[i] / 32;
        int elems = 27 * kcnt * 2 * 512;
        prep_weights_kernel<<<(elems + 255) / 256, 256, 0, stream>>>(
            cw[i], WGT[i], CinA[i], CoutA[i], kcnt);
        prep_bias_kernel<<<1, 32, 0, stream>>>(cb[i], BIA[i], CoutA[i]);
    }
    zero_bf16_kernel<<<1, 128, 0, stream>>>(BUF_IN  + zIN,  128);
    zero_bf16_kernel<<<1, 128, 0, stream>>>(BUF_T0A + z64,  128);
    zero_bf16_kernel<<<1, 128, 0, stream>>>(BUF_T0B + z64,  128);
    zero_bf16_kernel<<<1, 128, 0, stream>>>(BUF_R1A + zR1A, 128);
    zero_bf16_kernel<<<1, 128, 0, stream>>>(BUF_R1B + zR1B, 128);
    zero_bf16_kernel<<<1, 128, 0, stream>>>(BUF_R2A + zR2A, 128);
    zero_bf16_kernel<<<1, 128, 0, stream>>>(BUF_R2B + zR2B, 128);
    zero_bf16_kernel<<<1, 128, 0, stream>>>(BUF_R3A + zR3A, 128);
    zero_bf16_kernel<<<1, 128, 0, stream>>>(BUF_R3B + zR3B, 128);
    pack_input_kernel<<<(int)((R0 * 32 + 255) / 256), 256, 0, stream>>>(vol, BUF_IN, R0);

    auto conv = [&](int i, const __bf16* src, __bf16* dst,
                    int D, int H, int W, int Do, int Ho, int Wo,
                    int sC, int sO, int dC, int dO, int stride, unsigned zoff) {
        int nvox = Do * Ho * Wo;
        int blocks = (nvox + 255) / 256;                 // 64 voxels/wave, 4 waves
        if (CinP[i] == 32 && sC == 32)
            conv3_wmma_kernel<1, 32><<<blocks, 128, 0, stream>>>(
                src, dst, WGT[i], BIA[i], D, H, W, Do, Ho, Wo, sO, dC, dO, stride, zoff);
        else if (CinP[i] == 32)
            conv3_wmma_kernel<1, 64><<<blocks, 128, 0, stream>>>(
                src, dst, WGT[i], BIA[i], D, H, W, Do, Ho, Wo, sO, dC, dO, stride, zoff);
        else
            conv3_wmma_kernel<2, 64><<<blocks, 128, 0, stream>>>(
                src, dst, WGT[i], BIA[i], D, H, W, Do, Ho, Wo, sO, dC, dO, stride, zoff);
    };
    auto upsB = [&](const __bf16* src, __bf16* dst, int Ds, int Hs, int Ws,
                    int sC, int sO, int dC, int dO) {
        long total = (long)(8 * Ds * Hs * Ws) * 32;
        upsample_bf16_kernel<<<(int)((total + 255) / 256), 256, 0, stream>>>(
            src, dst, Ds, Hs, Ws, sC, sO, dC, dO, 32);
    };

    // ---- munet ----
    conv(0,  BUF_IN,  BUF_T0B, D0,H0,W0, D0,H0,W0, 32, 0, 64,32, 1, zIN);  // x1
    conv(1,  BUF_T0B, BUF_T0A, D0,H0,W0, D0,H0,W0, 64,32, 64,32, 1, z64);  // x2
    conv(2,  BUF_T0A, BUF_R1A, D0,H0,W0, D1,H1,W1, 64,32, 64,32, 2, z64);  // x3
    conv(3,  BUF_R1A, BUF_R2A, D1,H1,W1, D2,H2,W2, 64,32, 64,32, 2, zR1A); // x4
    conv(4,  BUF_R2A, BUF_R3A, D2,H2,W2, D3,H3,W3, 64,32, 64,32, 2, zR2A); // x5
    conv(5,  BUF_R3A, BUF_R3A, D3,H3,W3, D3,H3,W3, 64,32, 64, 0, 1, zR3A); // x6
    conv(6,  BUF_R3A, BUF_R3B, D3,H3,W3, D3,H3,W3, 64, 0, 32, 0, 1, zR3A); // d5
    upsB(BUF_R3B, BUF_R2A, D3,H3,W3, 32,0, 64,0);                          // up(d5)
    conv(7,  BUF_R2A, BUF_R2B, D2,H2,W2, D2,H2,W2, 64, 0, 32, 0, 1, zR2A); // d4
    flow_conv_kernel<<<(int)((R2 + 255) / 256), 256, 0, stream>>>(         // flow1(d4)
        BUF_R2B, F_R2, fw[0], fb[0], D2, H2, W2, 32, 0, 32);
    upsample3_f32_kernel<<<(int)((3 * R1 + 255) / 256), 256, 0, stream>>>(F_R2, F_T1, D2,H2,W2);
    upsample3_f32_kernel<<<(int)((3 * R0 + 255) / 256), 256, 0, stream>>>(F_T1, SVF[0], D1,H1,W1);
    upsB(BUF_R2B, BUF_R1A, D2,H2,W2, 32,0, 64,0);                          // up(d4)
    conv(8,  BUF_R1A, BUF_R1B, D1,H1,W1, D1,H1,W1, 64, 0, 32, 0, 1, zR1A); // d3
    flow_conv_kernel<<<(int)((R1 + 255) / 256), 256, 0, stream>>>(         // flow2(d3)
        BUF_R1B, F_R1, fw[1], fb[1], D1, H1, W1, 32, 0, 32);
    upsample3_f32_kernel<<<(int)((3 * R0 + 255) / 256), 256, 0, stream>>>(F_R1, SVF[1], D1,H1,W1);
    upsB(BUF_R1B, BUF_T0A, D1,H1,W1, 32,0, 64,0);                          // up(d3)
    conv(9,  BUF_T0A, BUF_T0B, D0,H0,W0, D0,H0,W0, 64, 0, 64, 0, 1, z64);  // d2
    flow_conv_kernel<<<(int)((R0 + 255) / 256), 256, 0, stream>>>(         // flow3(d2)
        BUF_T0B, SVF[2], fw[2], fb[2], D0, H0, W0, 64, 0, 32);
    conv(10, BUF_T0B, BUF_IN,  D0,H0,W0, D0,H0,W0, 64, 0, 32, 0, 1, z64);  // d1
    flow_conv_kernel<<<(int)((R0 + 255) / 256), 256, 0, stream>>>(         // flow4(d1)
        BUF_IN, SVF[3], fw[3], fb[3], D0, H0, W0, 32, 0, 16);

    // ---- per-SVF integration + smoothing + vertex update ----
    const float invpow = 1.0f / 128.0f;  // 1 / 2^NSTEPS
    const float* vprev = vert;
    for (int k = 0; k < 4; ++k) {
        scale_kernel<<<(int)((3 * R0 + 255) / 256), 256, 0, stream>>>(
            SVF[k], FLOWA, 3 * R0, invpow);
        float* a = FLOWA; float* b = FLOWB;
        for (int s = 0; s < NSTEPS; ++s) {
            warp_step_kernel<<<(int)((R0 + 255) / 256), 256, 0, stream>>>(a, b, D0, H0, W0);
            float* t = a; a = b; b = t;
        }
        gauss3_kernel<<<(int)((3 * R0 + 255) / 256), 256, 0, stream>>>(a, PHI, D0, H0, W0);
        float* vnext = (k == 3) ? (float*)d_out : ((k & 1) ? VB1 : VB0);
        interp_update_kernel<<<(NV + 255) / 256, 256, 0, stream>>>(
            vprev, PHI, vnext, NV, D0, H0, W0);
        vprev = vnext;
    }
}
